// FastAttention__16381005266988
// MI455X (gfx1250) — compile-verified, matrix operand order checked by reading, not run on hardware
//
#include <hip/hip_runtime.h>
#include <stdint.h>

// Performer causal linear attention for MI455X (gfx1250, wave32, WMMA).
// 3-pass chunk-parallel formulation; bf16 WMMA with f32 accumulation.

typedef __attribute__((ext_vector_type(16))) __bf16 v16bf;
typedef __attribute__((ext_vector_type(4)))  __bf16 v4bf;
typedef __attribute__((ext_vector_type(8)))  float  v8f;

#define CHUNK   128
#define DHEAD   64
#define NCHUNK  32
#define NBH     16      // B*H
#define SEQ     4096
#define EPS     1e-10f

// ws layout (floats):
//   states: [NBH][2 branches][NCHUNK][64*64]  -> 4,194,304 floats (16 MiB)
//   ksums : [NBH][2 branches][NCHUNK][64]     ->    65,536 floats (256 KiB)
#define WS_STATE_ELEMS ((size_t)NBH * 2 * NCHUNK * 4096)

__device__ __forceinline__ v8f wmma_bf16(v16bf a, v16bf b, v8f c) {
  // D = A(16x32) * B(32x16) + C ; emits v_wmma_f32_16x16x32_bf16
  return __builtin_amdgcn_wmma_f32_16x16x32_bf16(
      false, a, false, b, (short)0, c, false, false);
}

// Load one 16x32 bf16 fragment from row-major LDS [rows x stride].
// Works as the A operand (lane = matrix row) and, by layout symmetry, as the
// B operand for products of the form A * X^T when X is stored row-major
// (lane = output column = row of X).
// Per-lane K pattern (ISA 7.12.2): K = 16*(e/8) + 8*(lane/16) + (e%8)
//  -> two contiguous 8-element (16B) runs per lane (ds_load_b128 x2).
__device__ __forceinline__ v16bf load_frag(const __bf16* base, int stride,
                                           int rowbase, int kbase, int lane) {
  const int half = (lane >> 4) & 1;
  const int rc   = lane & 15;
  const __bf16* p = base + (size_t)(rowbase + rc) * stride + kbase + 8 * half;
  v16bf f;
#pragma unroll
  for (int e = 0; e < 8; ++e) f[e] = p[e];
#pragma unroll
  for (int e = 0; e < 8; ++e) f[8 + e] = p[16 + e];
  return f;
}

// ---------------------------------------------------------------------------
// Pass 1: per-chunk state S_c = K^T V (64x64, f32) for both branches,
//         plus per-chunk column sums of K (denominator feed).
// grid = NBH*NCHUNK blocks, 256 threads (8 waves).
// ---------------------------------------------------------------------------
__global__ __launch_bounds__(256) void pa_chunk_state(
    const float* __restrict__ k, const float* __restrict__ k_rot,
    const float* __restrict__ v, float* __restrict__ ws) {
  __shared__ __bf16 sKT[64 * 128];   // K^T : row d, col j   (16 KB)
  __shared__ __bf16 sVT[64 * 128];   // V^T : row e, col j   (16 KB)

  const int bh   = blockIdx.x >> 5;
  const int c    = blockIdx.x & 31;
  const int tid  = threadIdx.x;
  const int lane = tid & 31;
  const int wave = tid >> 5;

  const size_t inBase = (size_t)bh * SEQ * DHEAD + (size_t)c * CHUNK * DHEAD;
  float* stateWS = ws;
  float* ksumWS  = ws + WS_STATE_ELEMS;

  // Stage V^T once (vector global loads, transposed scalar LDS stores).
  {
    const float4* v4 = (const float4*)(v + inBase);
    for (int x = tid; x < CHUNK * DHEAD / 4; x += 256) {
      float4 t = v4[x];
      int lin = x * 4, i = lin >> 6, d = lin & 63;
      sVT[(d + 0) * 128 + i] = (__bf16)t.x;
      sVT[(d + 1) * 128 + i] = (__bf16)t.y;
      sVT[(d + 2) * 128 + i] = (__bf16)t.z;
      sVT[(d + 3) * 128 + i] = (__bf16)t.w;
    }
  }

  for (int br = 0; br < 2; ++br) {
    const float* kin = br ? k_rot : k;
    __syncthreads();                       // prior-iteration readers done
    {
      const float4* k4 = (const float4*)(kin + inBase);
      for (int x = tid; x < CHUNK * DHEAD / 4; x += 256) {
        float4 t = k4[x];
        int lin = x * 4, i = lin >> 6, d = lin & 63;
        sKT[(d + 0) * 128 + i] = (__bf16)t.x;
        sKT[(d + 1) * 128 + i] = (__bf16)t.y;
        sKT[(d + 2) * 128 + i] = (__bf16)t.z;
        sKT[(d + 3) * 128 + i] = (__bf16)t.w;
      }
    }
    __syncthreads();

    // colsum_j K[j][d] == rowsum of K^T row d
    float* ksum = ksumWS + (((size_t)bh * 2 + br) * NCHUNK + c) * 64;
    if (tid < 64) {
      float s = 0.f;
      for (int j = 0; j < 128; ++j) s += (float)sKT[tid * 128 + j];
      ksum[tid] = s;
    }

    // S = K^T V : 4x4 tiles of 16x16, K-dim = 128 -> 4 wmma steps per tile.
    float* S = stateWS + (((size_t)bh * 2 + br) * NCHUNK + c) * 4096;
#pragma unroll
    for (int t = 0; t < 2; ++t) {          // 2 tiles per wave
      int tile = wave * 2 + t;
      int dt = tile >> 2, et = tile & 3;
      v8f acc = {0.f, 0.f, 0.f, 0.f, 0.f, 0.f, 0.f, 0.f};
#pragma unroll
      for (int ks = 0; ks < 4; ++ks) {
        v16bf a = load_frag(sKT, 128, 16 * dt, 32 * ks, lane);  // A = K^T
        v16bf b = load_frag(sVT, 128, 16 * et, 32 * ks, lane);  // B = V
        acc = wmma_bf16(a, b, acc);
      }
      const int hi = (lane >> 4) & 1, n = lane & 15;
#pragma unroll
      for (int r = 0; r < 8; ++r) {
        int d = 16 * dt + r + 8 * hi;
        int e = 16 * et + n;
        S[d * 64 + e] = acc[r];
      }
    }
  }
}

// ---------------------------------------------------------------------------
// Pass 2: in-place EXCLUSIVE prefix sum over the 32 chunks, per (bh, branch),
// for both the 64x64 states and the 64-wide ksum vectors.
// grid = NBH*2 blocks, 256 threads. Deterministic per-thread serial scan.
// ---------------------------------------------------------------------------
__global__ __launch_bounds__(256) void pa_prefix(float* __restrict__ ws) {
  const int g   = blockIdx.x;   // bh*2 + branch
  const int tid = threadIdx.x;
  float* stateWS = ws;
  float* ksumWS  = ws + WS_STATE_ELEMS;

  for (int idx = tid; idx < 4096; idx += 256) {
    float run = 0.f;
    for (int c = 0; c < NCHUNK; ++c) {
      float* p = stateWS + ((size_t)g * NCHUNK + c) * 4096 + idx;
      float t = *p; *p = run; run += t;
    }
  }
  if (tid < 64) {
    float run = 0.f;
    for (int c = 0; c < NCHUNK; ++c) {
      float* p = ksumWS + ((size_t)g * NCHUNK + c) * 64 + tid;
      float t = *p; *p = run; run += t;
    }
  }
}

// ---------------------------------------------------------------------------
// Pass 3: per chunk, fully parallel:
//   scores  = Q K^T + Qr Kr^T            (branches fuse: shared mask & V)
//   drow_i  = rowsum(mask . scores)_i    (intra denominator)
//   dint_i  = q_i.ksumP + qr_i.krsumP    (inter denominator, via WMMA against
//                                         a column-broadcast ksum operand)
//   num     = (mask . scores) V + Q P + Qr Pr
//   out     = num / (drow + dint + EPS)
// grid = NBH*NCHUNK blocks, 256 threads (8 waves; wave w owns rows 16w..16w+15)
// ---------------------------------------------------------------------------
__global__ __launch_bounds__(256) void pa_output(
    const float* __restrict__ q, const float* __restrict__ k,
    const float* __restrict__ q_rot, const float* __restrict__ k_rot,
    const float* __restrict__ v, const float* __restrict__ ws,
    float* __restrict__ out) {
  __shared__ __bf16 sA[128 * 128];   // phase 1-2: Q (stride 64) | K (stride 64)
                                     // phase 3  : masked scores (stride 128)  32 KB
  __shared__ __bf16 sVT[64 * 128];   // V^T                                    16 KB
  __shared__ __bf16 sPT[64 * 64];    // P^T (per-branch, reused)                8 KB
  __shared__ __bf16 sKSB[16 * 64];   // ksum broadcast: 16 identical rows       2 KB

  const int bh   = blockIdx.x >> 5;
  const int c    = blockIdx.x & 31;
  const int tid  = threadIdx.x;
  const int lane = tid & 31;
  const int wave = tid >> 5;
  const int hi   = (lane >> 4) & 1;
  const int n    = lane & 15;

  const size_t inBase = (size_t)bh * SEQ * DHEAD + (size_t)c * CHUNK * DHEAD;
  const float* stateWS = ws;
  const float* ksumWS  = ws + WS_STATE_ELEMS;

  // Stage V^T once.
  {
    const float4* v4 = (const float4*)(v + inBase);
    for (int x = tid; x < CHUNK * DHEAD / 4; x += 256) {
      float4 t = v4[x];
      int lin = x * 4, i = lin >> 6, d = lin & 63;
      sVT[(d + 0) * 128 + i] = (__bf16)t.x;
      sVT[(d + 1) * 128 + i] = (__bf16)t.y;
      sVT[(d + 2) * 128 + i] = (__bf16)t.z;
      sVT[(d + 3) * 128 + i] = (__bf16)t.w;
    }
  }

  v8f sacc[8], oacc[4], dacc;
  const v8f vzero = {0.f, 0.f, 0.f, 0.f, 0.f, 0.f, 0.f, 0.f};
#pragma unroll
  for (int t = 0; t < 8; ++t) sacc[t] = vzero;
#pragma unroll
  for (int t = 0; t < 4; ++t) oacc[t] = vzero;
  dacc = vzero;

  for (int br = 0; br < 2; ++br) {
    const float* qin = br ? q_rot : q;
    const float* kin = br ? k_rot : k;
    // --- stage branch operands (vector loads; contiguous packed stores) ---
    {
      const float4* q4 = (const float4*)(qin + inBase);
      const float4* k4 = (const float4*)(kin + inBase);
      for (int x = tid; x < CHUNK * DHEAD / 4; x += 256) {
        float4 tq = q4[x];
        float4 tk = k4[x];
        v4bf bq = {(__bf16)tq.x, (__bf16)tq.y, (__bf16)tq.z, (__bf16)tq.w};
        v4bf bk = {(__bf16)tk.x, (__bf16)tk.y, (__bf16)tk.z, (__bf16)tk.w};
        *(v4bf*)(sA + 4 * x)            = bq;   // Q rows, stride 64
        *(v4bf*)(sA + 128 * 64 + 4 * x) = bk;   // K rows, stride 64
      }
    }
    {
      const float4* P4 = (const float4*)(stateWS +
          (((size_t)bh * 2 + br) * NCHUNK + c) * 4096);
      for (int x = tid; x < 64 * 64 / 4; x += 256) {
        float4 t = P4[x];
        int lin = x * 4, d = lin >> 6, e = lin & 63;
        sPT[(e + 0) * 64 + d] = (__bf16)t.x;   // P^T
        sPT[(e + 1) * 64 + d] = (__bf16)t.y;
        sPT[(e + 2) * 64 + d] = (__bf16)t.z;
        sPT[(e + 3) * 64 + d] = (__bf16)t.w;
      }
    }
    {
      const float* ksum = ksumWS + (((size_t)bh * 2 + br) * NCHUNK + c) * 64;
      for (int x = tid; x < 16 * 64; x += 256)
        sKSB[x] = (__bf16)ksum[x & 63];        // 16 identical rows
    }
    __syncthreads();

    const __bf16* Qs = sA;
    const __bf16* Ks = sA + 128 * 64;
    // scores += Q K^T  (this wave's 16-row strip x all 128 cols)
#pragma unroll
    for (int ct = 0; ct < 8; ++ct)
#pragma unroll
      for (int ks = 0; ks < 2; ++ks)
        sacc[ct] = wmma_bf16(load_frag(Qs, 64, 16 * wave, 32 * ks, lane),
                             load_frag(Ks, 64, 16 * ct,   32 * ks, lane),
                             sacc[ct]);
    // inter numerator += Q P
#pragma unroll
    for (int ct = 0; ct < 4; ++ct)
#pragma unroll
      for (int ks = 0; ks < 2; ++ks)
        oacc[ct] = wmma_bf16(load_frag(Qs,  64, 16 * wave, 32 * ks, lane),
                             load_frag(sPT, 64, 16 * ct,   32 * ks, lane),
                             oacc[ct]);
    // inter denominator += q_i . ksum : B = column-broadcast ksum, so every
    // column of the result tile holds the row's dot product (right lanes,
    // right registers, no shuffles).
#pragma unroll
    for (int ks = 0; ks < 2; ++ks)
      dacc = wmma_bf16(load_frag(Qs,   64, 16 * wave, 32 * ks, lane),
                       load_frag(sKSB, 64, 0,         32 * ks, lane),
                       dacc);
    __syncthreads();   // done reading this branch's LDS
  }

  // --- causal mask, rowsum, stash bf16 scores over the Q/K region ---
  float drow[8];
#pragma unroll
  for (int r = 0; r < 8; ++r) drow[r] = 0.f;
#pragma unroll
  for (int ct = 0; ct < 8; ++ct)
#pragma unroll
    for (int r = 0; r < 8; ++r) {
      int i = 16 * wave + r + 8 * hi;       // C-layout: row = reg + 8*(lane>=16)
      int j = 16 * ct + n;                  //            col = lane % 16
      float s = (j <= i) ? sacc[ct][r] : 0.f;
      drow[r] += s;
      sA[i * 128 + j] = (__bf16)s;
    }
#pragma unroll
  for (int m = 8; m >= 1; m >>= 1)
#pragma unroll
    for (int r = 0; r < 8; ++r) drow[r] += __shfl_xor(drow[r], m, 32);
  __syncthreads();

  // --- intra numerator += (mask . scores) V ---
#pragma unroll
  for (int ct = 0; ct < 4; ++ct)
#pragma unroll
    for (int ks = 0; ks < 4; ++ks)
      oacc[ct] = wmma_bf16(load_frag(sA,  128, 16 * wave, 32 * ks, lane),
                           load_frag(sVT, 128, 16 * ct,   32 * ks, lane),
                           oacc[ct]);

  // --- divide & store ---
#pragma unroll
  for (int ct = 0; ct < 4; ++ct)
#pragma unroll
    for (int r = 0; r < 8; ++r) {
      int i = 16 * wave + r + 8 * hi;
      int e = 16 * ct + n;
      float den = drow[r] + dacc[r] + EPS;
      out[inBase + (size_t)i * DHEAD + e] = oacc[ct][r] / den;
    }
}

extern "C" void kernel_launch(void* const* d_in, const int* in_sizes, int n_in,
                              void* d_out, int out_size, void* d_ws, size_t ws_size,
                              hipStream_t stream) {
  const float* q     = (const float*)d_in[0];
  const float* k     = (const float*)d_in[1];
  const float* q_rot = (const float*)d_in[2];
  const float* k_rot = (const float*)d_in[3];
  const float* v     = (const float*)d_in[4];
  float* out = (float*)d_out;
  float* ws  = (float*)d_ws;   // needs 16 MiB + 256 KiB

  dim3 blk(256);
  pa_chunk_state<<<dim3(NBH * NCHUNK), blk, 0, stream>>>(k, k_rot, v, ws);
  pa_prefix     <<<dim3(NBH * 2),      blk, 0, stream>>>(ws);
  pa_output     <<<dim3(NBH * NCHUNK), blk, 0, stream>>>(q, k, q_rot, k_rot, v, ws, out);
}
